// CorrectTransformerAdaptor_33870112096590
// MI455X (gfx1250) — matrix operand order repaired and checked by reading, not compiled, not run on hardware
//
#include <hip/hip_runtime.h>

// ---------------------------------------------------------------------------
// CDNA5 (gfx1250) transformer adaptor: bf16 WMMA GEMMs with async-to-LDS
// double-buffered staging + flash attention. Residual stream f32 in d_out.
// ---------------------------------------------------------------------------

typedef __attribute__((ext_vector_type(16))) __bf16 v16bf;
typedef __attribute__((ext_vector_type(8)))  __bf16 v8bf;
typedef __attribute__((ext_vector_type(8)))  float  v8f;

static __device__ inline v8f wmma_bf16(v16bf a, v16bf b, v8f c) {
  // D(16x16 f32) = A(16x32 bf16) * B(32x16 bf16) + C
  return __builtin_amdgcn_wmma_f32_16x16x32_bf16(
      false, a, false, b, (short)0, c, false, false);
}

// A-fragment: 16x32 bf16 tile, row-major, lane m = lane&15, half g = lane>>4
// elements 0..7 -> K = g*8 + e ; elements 8..15 -> K = 16 + g*8 + e
static __device__ inline v16bf load_a(const __bf16* p, int ld) {
  const int lane = threadIdx.x & 31;
  const int m = lane & 15, g = lane >> 4;
  const __bf16* q = p + (long)m * ld + g * 8;
  union { v16bf v; v8bf h[2]; } u;
  u.h[0] = *(const v8bf*)(q);
  u.h[1] = *(const v8bf*)(q + 16);
  return u.v;
}

// B-fragment: 32x16 bf16 tile where storage is Bt[N][K] row-major (N=lane&15,
// half g: elements 0..7 -> K = g*8 + e ; elements 8..15 -> K = 16 + g*8 + e).
static __device__ inline v16bf load_b(const __bf16* p, int ld) {
  const int lane = threadIdx.x & 31;
  const int n = lane & 15, g = lane >> 4;
  const __bf16* q = p + (long)n * ld + g * 8;
  union { v16bf v; v8bf h[2]; } u;
  u.h[0] = *(const v8bf*)(q);
  u.h[1] = *(const v8bf*)(q + 16);
  return u.v;
}

// Low 32 bits of a generic pointer to __shared__ == LDS byte offset.
static __device__ inline unsigned lds_off(const void* p) {
  return (unsigned)(unsigned long long)(uintptr_t)p;
}

// CDNA5 async copy: LDS[loff] = MEM[gaddr], 16B per lane, tracked by ASYNCcnt.
static __device__ inline void async_load16(unsigned loff, const void* g) {
  unsigned long long ga = (unsigned long long)(uintptr_t)g;
  asm volatile("global_load_async_to_lds_b128 %0, %1, off"
               :: "v"(loff), "v"(ga) : "memory");
}
static __device__ inline void wait_async0() {
  asm volatile("s_wait_asynccnt 0" ::: "memory");
}

static __device__ inline float hmax16(float v) {
  v = fmaxf(v, __shfl_xor(v, 1, 16));
  v = fmaxf(v, __shfl_xor(v, 2, 16));
  v = fmaxf(v, __shfl_xor(v, 4, 16));
  v = fmaxf(v, __shfl_xor(v, 8, 16));
  return v;
}
static __device__ inline float hsum16(float v) {
  v += __shfl_xor(v, 1, 16);
  v += __shfl_xor(v, 2, 16);
  v += __shfl_xor(v, 4, 16);
  v += __shfl_xor(v, 8, 16);
  return v;
}

// ---------------------------------------------------------------------------
// f32 -> bf16 elementwise convert
// ---------------------------------------------------------------------------
__global__ void cvt_bf16_k(const float* __restrict__ s, __bf16* __restrict__ d, int n) {
  int i = blockIdx.x * 256 + threadIdx.x;
  if (i < n) d[i] = (__bf16)s[i];
}

// ---------------------------------------------------------------------------
// GEMM: C[M,N] = epi(A[M,K] @ Bt[N,K]^T + bias[N])   (M fixed 8192 via grid.y)
// 256 thr / 8 waves; block tile 128x128; wave tile 32x64; K-step 32.
// A and B block tiles double-buffered in LDS via global_load_async_to_lds.
// ---------------------------------------------------------------------------
#define LDS_STRIDE 40  // 32 + 8 pad elements -> conflict-free frag loads

template <bool RELU, bool RESID, bool OUTF, bool OUTB>
__global__ __launch_bounds__(256) void gemm_k(
    const __bf16* __restrict__ A, int lda,
    const __bf16* __restrict__ Bt, int ldb,
    const float* __restrict__ bias,
    const float* __restrict__ resid,
    float* __restrict__ outF,
    __bf16* __restrict__ outB,
    int N, int Kd)
{
  __shared__ __bf16 sA[2][128 * LDS_STRIDE];  // 2 x 10 KB
  __shared__ __bf16 sB[2][128 * LDS_STRIDE];  // 2 x 10 KB

  const int tid  = threadIdx.x;
  const int lane = tid & 31;
  const int wave = tid >> 5;
  const int blkm = blockIdx.y * 128;
  const int blkn = blockIdx.x * 128;
  const int wm   = (wave >> 1) * 32;
  const int wn   = (wave & 1) * 64;

  const __bf16* Ablk = A  + (long)blkm * lda;
  const __bf16* Bblk = Bt + (long)blkn * ldb;

  // Stage one 128x32 bf16 tile (64 B rows = 4 x 16 B chunks) into LDS.
  // 512 chunks total, 256 threads -> 2 async instructions per thread.
  auto stage = [&](const __bf16* src, int ld, __bf16* dst, int kt) {
#pragma unroll
    for (int r = 0; r < 2; ++r) {
      int c   = r * 256 + tid;
      int row = c >> 2, cw = c & 3;
      async_load16(lds_off(dst + row * LDS_STRIDE + cw * 8),
                   src + (long)row * ld + kt + cw * 8);
    }
  };

  stage(Ablk, lda, sA[0], 0);
  stage(Bblk, ldb, sB[0], 0);

  v8f acc[2][4] = {};
  const int nsteps = Kd >> 5;

  for (int s = 0; s < nsteps; ++s) {
    const int cur = s & 1;
    wait_async0();      // my async writes into sA/sB[cur] are in LDS
    __syncthreads();    // everyone's writes visible; prev buffer fully read

    if (s + 1 < nsteps) {
      stage(Ablk, lda, sA[cur ^ 1], (s + 1) * 32);
      stage(Bblk, ldb, sB[cur ^ 1], (s + 1) * 32);
    }

    const __bf16* At = sA[cur] + wm * LDS_STRIDE;
    const __bf16* Bw = sB[cur] + wn * LDS_STRIDE;
    v16bf a0 = load_a(At, LDS_STRIDE);
    v16bf a1 = load_a(At + 16 * LDS_STRIDE, LDS_STRIDE);
#pragma unroll
    for (int j = 0; j < 4; ++j) {
      v16bf bf = load_b(Bw + (j * 16) * LDS_STRIDE, LDS_STRIDE);
      acc[0][j] = wmma_bf16(a0, bf, acc[0][j]);
      acc[1][j] = wmma_bf16(a1, bf, acc[1][j]);
    }
  }

  const int nl = lane & 15, g = lane >> 4;
#pragma unroll
  for (int j = 0; j < 4; ++j) {
    int ncol = blkn + wn + j * 16 + nl;
    float bv = bias[ncol];
#pragma unroll
    for (int i = 0; i < 2; ++i) {
#pragma unroll
      for (int r = 0; r < 8; ++r) {
        long idx = (long)(blkm + wm + i * 16 + r + 8 * g) * N + ncol;
        float v = acc[i][j][r] + bv;
        if (RELU)  v = fmaxf(v, 0.0f);
        if (RESID) v += resid[idx];
        if (OUTF)  outF[idx] = v;
        if (OUTB)  outB[idx] = (__bf16)v;
      }
    }
  }
}

// ---------------------------------------------------------------------------
// LayerNorm over D=1024, one block per row, bf16 output.
// ---------------------------------------------------------------------------
__global__ __launch_bounds__(256) void ln_k(const float* __restrict__ X,
                                            const float* __restrict__ gamma,
                                            const float* __restrict__ beta,
                                            __bf16* __restrict__ out)
{
  __shared__ float red[256];
  const int row = blockIdx.x;
  const float* x = X + (long)row * 1024;
  const int t = threadIdx.x;

  float s = 0.f;
  for (int i = t; i < 1024; i += 256) s += x[i];
  red[t] = s; __syncthreads();
  for (int o = 128; o > 0; o >>= 1) { if (t < o) red[t] += red[t + o]; __syncthreads(); }
  float mean = red[0] * (1.0f / 1024.0f);
  __syncthreads();

  float vs = 0.f;
  for (int i = t; i < 1024; i += 256) { float d = x[i] - mean; vs += d * d; }
  red[t] = vs; __syncthreads();
  for (int o = 128; o > 0; o >>= 1) { if (t < o) red[t] += red[t + o]; __syncthreads(); }
  float inv = rsqrtf(red[0] * (1.0f / 1024.0f) + 1e-12f);

  for (int i = t; i < 1024; i += 256)
    out[(long)row * 1024 + i] = (__bf16)((x[i] - mean) * inv * gamma[i] + beta[i]);
}

// ---------------------------------------------------------------------------
// V [8192,1024] bf16 -> Vt [B,H,DK,T] bf16
// ---------------------------------------------------------------------------
__global__ void transpose_v_k(const __bf16* __restrict__ V, __bf16* __restrict__ Vt) {
  long idx = (long)blockIdx.x * 256 + threadIdx.x;  // 8*1024*1024 elems
  int t = (int)(idx & 1023);
  int d = (int)((idx >> 10) & 127);
  int h = (int)((idx >> 17) & 7);
  int b = (int)(idx >> 20);
  Vt[idx] = V[((long)(b * 1024 + t)) * 1024 + h * 128 + d];
}

// ---------------------------------------------------------------------------
// Flash attention: T=1024, H=8, DK=128.  grid(qblk=8, H=8, B=8), 8 waves/blk.
// Each wave: 16 q-rows, full 128-d output; online softmax via width-16
// shuffles; P re-layout (D->A frag) through a wave-private LDS slab.
// ---------------------------------------------------------------------------
__global__ __launch_bounds__(256) void attn_k(const __bf16* __restrict__ Q,
                                              const __bf16* __restrict__ Kt,
                                              const __bf16* __restrict__ Vt,
                                              __bf16* __restrict__ O)
{
  __shared__ __bf16 plds[8][512];  // 8 waves * 16x32 bf16 P tile
  const int lane = threadIdx.x & 31;
  const int wave = threadIdx.x >> 5;
  const int g = lane >> 4;
  const int nl = lane & 15;
  const int b = blockIdx.z, h = blockIdx.y;
  const int q0 = blockIdx.x * 128 + wave * 16;
  const float SC = 0.08838834764831845f;  // 1/sqrt(128)

  const __bf16* Qb = Q + ((long)(b * 1024 + q0)) * 1024 + h * 128;
  const __bf16* Kb = Kt + ((long)b * 1024) * 1024 + h * 128;
  const __bf16* Vb = Vt + ((long)(b * 8 + h)) * 128 * 1024;

  v16bf qf[4];
#pragma unroll
  for (int dc = 0; dc < 4; ++dc) qf[dc] = load_a(Qb + dc * 32, 1024);

  v8f oacc[8] = {};
  float rmax[8], rsum[8];
#pragma unroll
  for (int r = 0; r < 8; ++r) { rmax[r] = -3.0e38f; rsum[r] = 0.f; }

  for (int kt = 0; kt < 1024; kt += 32) {
    v8f s0 = {}, s1 = {};
#pragma unroll
    for (int dc = 0; dc < 4; ++dc) {
      v16bf k0 = load_b(Kb + (long)kt * 1024 + dc * 32, 1024);
      v16bf k1 = load_b(Kb + (long)(kt + 16) * 1024 + dc * 32, 1024);
      s0 = wmma_bf16(qf[dc], k0, s0);
      s1 = wmma_bf16(qf[dc], k1, s1);
    }
    // online softmax per row (row r+8g lives in reg r, lane-half g)
#pragma unroll
    for (int r = 0; r < 8; ++r) {
      float v0 = s0[r] * SC;
      float v1 = s1[r] * SC;
      float tm = hmax16(fmaxf(v0, v1));
      float nm = fmaxf(rmax[r], tm);
      float corr = __expf(rmax[r] - nm);
      float p0 = __expf(v0 - nm);
      float p1 = __expf(v1 - nm);
      float ps = hsum16(p0 + p1);
      rsum[r] = rsum[r] * corr + ps;
      rmax[r] = nm;
#pragma unroll
      for (int j = 0; j < 8; ++j) oacc[j][r] *= corr;
      int mr = r + 8 * g;
      plds[wave][mr * 32 + nl]      = (__bf16)p0;
      plds[wave][mr * 32 + 16 + nl] = (__bf16)p1;
    }
    // reload P as A-fragment (same wave: DS ops stay ordered)
    union { v16bf v; v8bf hh[2]; } u;
    const __bf16* pp = &plds[wave][nl * 32 + g * 8];
    u.hh[0] = *(const v8bf*)(pp);
    u.hh[1] = *(const v8bf*)(pp + 16);
    v16bf pa = u.v;
#pragma unroll
    for (int j = 0; j < 8; ++j) {
      v16bf vb = load_b(Vb + (long)(j * 16) * 1024 + kt, 1024);
      oacc[j] = wmma_bf16(pa, vb, oacc[j]);
    }
  }

#pragma unroll
  for (int j = 0; j < 8; ++j) {
    int col = h * 128 + j * 16 + nl;
#pragma unroll
    for (int r = 0; r < 8; ++r) {
      int row = q0 + r + 8 * g;
      O[((long)(b * 1024 + row)) * 1024 + col] = (__bf16)(oacc[j][r] / rsum[r]);
    }
  }
}

// ---------------------------------------------------------------------------
// Host orchestration.  M = 8192 rows throughout.  Needs ~146 MB workspace.
// ---------------------------------------------------------------------------
extern "C" void kernel_launch(void* const* d_in, const int* in_sizes, int n_in,
                              void* d_out, int out_size, void* d_ws, size_t ws_size,
                              hipStream_t stream)
{
  (void)in_sizes; (void)n_in; (void)out_size; (void)ws_size;

  const float* x    = (const float*)d_in[0];
  const float* W1   = (const float*)d_in[1];
  const float* b1   = (const float*)d_in[2];
  const float* W2   = (const float*)d_in[3];
  const float* b2   = (const float*)d_in[4];
  const float* ln1g = (const float*)d_in[5];
  const float* ln1b = (const float*)d_in[6];
  const float* ln2g = (const float*)d_in[7];
  const float* ln2b = (const float*)d_in[8];
  const float* Wq   = (const float*)d_in[9];
  const float* bq   = (const float*)d_in[10];
  const float* Wk   = (const float*)d_in[11];
  const float* bk   = (const float*)d_in[12];
  const float* Wv   = (const float*)d_in[13];
  const float* bv   = (const float*)d_in[14];
  const float* Wo   = (const float*)d_in[15];
  const float* bo   = (const float*)d_in[16];
  const float* Fw1  = (const float*)d_in[17];
  const float* Fb1  = (const float*)d_in[18];
  const float* Fw2  = (const float*)d_in[19];
  const float* Fb2  = (const float*)d_in[20];

  float* X = (float*)d_out;  // residual stream [8192,1024] f32
  char* ws = (char*)d_ws;
  const size_t MB = 1024 * 1024;

  __bf16* XBF = (__bf16*)(ws + 0 * MB);    // 16 MB
  __bf16* HBF = (__bf16*)(ws + 16 * MB);   // 16 MB  (LN output)
  __bf16* H1  = (__bf16*)(ws + 32 * MB);   // 32 MB  (downsample hidden)
  __bf16* Qb  = (__bf16*)(ws + 32 * MB);   // reuse H1 region after downsample
  __bf16* Kb  = (__bf16*)(ws + 48 * MB);
  __bf16* Vb  = (__bf16*)(ws + 64 * MB);   // 16 MB
  __bf16* VT  = (__bf16*)(ws + 80 * MB);   // 16 MB
  __bf16* Ob  = (__bf16*)(ws + 96 * MB);   // 16 MB
  __bf16* FF  = (__bf16*)(ws + 112 * MB);  // 4 MB
  __bf16* W1b = (__bf16*)(ws + 120 * MB);  // 4 MB
  __bf16* W2b = (__bf16*)(ws + 124 * MB);  // 4 MB
  __bf16* Wqb = (__bf16*)(ws + 128 * MB);  // 4 MB (2 layers)
  __bf16* Wkb = (__bf16*)(ws + 132 * MB);
  __bf16* Wvb = (__bf16*)(ws + 136 * MB);
  __bf16* Wob = (__bf16*)(ws + 140 * MB);
  __bf16* F1b = (__bf16*)(ws + 144 * MB);  // 1 MB (2 layers)
  __bf16* F2b = (__bf16*)(ws + 145 * MB);  // 1 MB

  auto cvt = [&](const float* s, __bf16* d, long n) {
    cvt_bf16_k<<<dim3((unsigned)((n + 255) / 256)), dim3(256), 0, stream>>>(s, d, (int)n);
  };
  cvt(x,   XBF, 8192L * 1024);
  cvt(W1,  W1b, 2048L * 1024);
  cvt(W2,  W2b, 1024L * 2048);
  cvt(Wq,  Wqb, 2L * 1024 * 1024);
  cvt(Wk,  Wkb, 2L * 1024 * 1024);
  cvt(Wv,  Wvb, 2L * 1024 * 1024);
  cvt(Wo,  Wob, 2L * 1024 * 1024);
  cvt(Fw1, F1b, 2L * 256 * 1024);
  cvt(Fw2, F2b, 2L * 1024 * 256);

  // ---- downsample MLP ----
  gemm_k<true, false, false, true><<<dim3(16, 64), 256, 0, stream>>>(
      XBF, 1024, W1b, 1024, b1, nullptr, nullptr, H1, 2048, 1024);
  gemm_k<false, false, true, false><<<dim3(8, 64), 256, 0, stream>>>(
      H1, 2048, W2b, 2048, b2, nullptr, X, nullptr, 1024, 2048);

  // ---- transformer layers ----
  for (int l = 0; l < 2; ++l) {
    const long wOff = (long)l * 1024 * 1024;
    ln_k<<<8192, 256, 0, stream>>>(X, ln1g + l * 1024, ln1b + l * 1024, HBF);

    gemm_k<false, false, false, true><<<dim3(8, 64), 256, 0, stream>>>(
        HBF, 1024, Wqb + wOff, 1024, bq + l * 1024, nullptr, nullptr, Qb, 1024, 1024);
    gemm_k<false, false, false, true><<<dim3(8, 64), 256, 0, stream>>>(
        HBF, 1024, Wkb + wOff, 1024, bk + l * 1024, nullptr, nullptr, Kb, 1024, 1024);
    gemm_k<false, false, false, true><<<dim3(8, 64), 256, 0, stream>>>(
        HBF, 1024, Wvb + wOff, 1024, bv + l * 1024, nullptr, nullptr, Vb, 1024, 1024);

    transpose_v_k<<<32768, 256, 0, stream>>>(Vb, VT);
    attn_k<<<dim3(8, 8, 8), 256, 0, stream>>>(Qb, Kb, VT, Ob);

    gemm_k<false, true, true, false><<<dim3(8, 64), 256, 0, stream>>>(
        Ob, 1024, Wob + wOff, 1024, bo + l * 1024, X, X, nullptr, 1024, 1024);

    ln_k<<<8192, 256, 0, stream>>>(X, ln2g + l * 1024, ln2b + l * 1024, HBF);

    gemm_k<true, false, false, true><<<dim3(2, 64), 256, 0, stream>>>(
        HBF, 1024, F1b + (long)l * 256 * 1024, 1024, Fb1 + l * 256,
        nullptr, nullptr, FF, 256, 1024);
    gemm_k<false, true, true, false><<<dim3(8, 64), 256, 0, stream>>>(
        FF, 256, F2b + (long)l * 1024 * 256, 256, Fb2 + l * 1024, X, X, nullptr, 1024, 256);
  }
}
